// MambaBlock_20822001451658
// MI455X (gfx1250) — compile-verified, matrix operand order checked by reading, not run on hardware
//
#include <hip/hip_runtime.h>

#define DIM   128
#define DST   16
#define EE    256
#define BSZ   16
#define LLEN  4096   // H*W
#define LN_EPS 1e-5f

typedef __attribute__((ext_vector_type(16))) __bf16       v16bf;
typedef __attribute__((ext_vector_type(8)))  float        v8f;
typedef __attribute__((ext_vector_type(8)))  unsigned int v8u;
typedef __attribute__((ext_vector_type(2)))  float        v2f;

// K-offset inside a 32-wide K block for the bf16 WMMA A/B fragment layout
// (pair p = register half r; lanes 16-31 hold K+8 within each 16-K half).
__device__ __forceinline__ int kpair_base(int p, int lane) {
    return (p >> 2) * 16 + ((lane & 16) ? 8 : 0) + (p & 3) * 2;
}

// ---------------------------------------------------------------------------
// Kernel 1: fold the E=256 dimension into small fused weights, then pack the
// bf16 copies directly in WMMA B-fragment order.
// ---------------------------------------------------------------------------
__global__ __launch_bounds__(256) void k_pre(
    const float* __restrict__ Bm, const float* __restrict__ Cm,
    const float* __restrict__ D,  const float* __restrict__ W_in,
    const float* __restrict__ b_in, const float* __restrict__ W_out,
    const float* __restrict__ b_out,
    float* __restrict__ Wu, float* __restrict__ Mg, float* __restrict__ WC,
    float* __restrict__ cb, float* __restrict__ bu,
    __bf16* __restrict__ Mbf, __bf16* __restrict__ Wubf)
{
    int tid = threadIdx.x;
    // M[c1][c2] = sum_e W_in[e][c1] * D[e] * W_out[c2][e]
    for (int o = tid; o < DIM * DIM; o += 256) {
        int c1 = o >> 7, c2 = o & 127;
        float acc = 0.f;
        for (int e = 0; e < EE; ++e)
            acc += W_in[e * DIM + c1] * D[e] * W_out[c2 * EE + e];
        Mg[o] = acc;
    }
    // Wu[s][c] = sum_e Bm[s][e] * W_in[e][c]
    for (int o = tid; o < DST * DIM; o += 256) {
        int s = o >> 7, c = o & 127;
        float acc = 0.f;
        for (int e = 0; e < EE; ++e)
            acc += Bm[s * EE + e] * W_in[e * DIM + c];
        Wu[o] = acc;
    }
    // WC[c][s] = sum_e W_out[c][e] * Cm[e][s]
    for (int o = tid; o < DIM * DST; o += 256) {
        int c = o >> 4, s = o & 15;
        float acc = 0.f;
        for (int e = 0; e < EE; ++e)
            acc += W_out[c * EE + e] * Cm[e * DST + s];
        WC[o] = acc;
    }
    if (tid < DIM) {
        float acc = b_out[tid];
        for (int e = 0; e < EE; ++e) acc += b_in[e] * D[e] * W_out[tid * EE + e];
        cb[tid] = acc;
    }
    if (tid < DST) {
        float acc = 0.f;
        for (int e = 0; e < EE; ++e) acc += Bm[tid * EE + e] * b_in[e];
        bu[tid] = acc;
    }
    __threadfence();
    __syncthreads();
    // Pack Mbf: [nb(8)][kb(4)][lane(32)][j(16)]; B[k][n] = M[k][nb*16+n]
    for (int idx = tid; idx < 8 * 4 * 32; idx += 256) {
        int nb = idx >> 7, kb = (idx >> 5) & 3, lane = idx & 31;
        for (int j = 0; j < 16; ++j) {
            int K = kb * 32 + kpair_base(j >> 1, lane) + (j & 1);
            int N = nb * 16 + (lane & 15);
            Mbf[(size_t)idx * 16 + j] = (__bf16)Mg[K * DIM + N];
        }
    }
    // Pack Wubf: [kb(4)][lane(32)][j(16)]; B[k][n] = Wu[n][k]
    for (int idx = tid; idx < 4 * 32; idx += 256) {
        int kb = idx >> 5, lane = idx & 31;
        for (int j = 0; j < 16; ++j) {
            int K = kb * 32 + kpair_base(j >> 1, lane) + (j & 1);
            int N = lane & 15;
            Wubf[idx * 16 + j] = (__bf16)Wu[N * DIM + K];
        }
    }
}

// ---------------------------------------------------------------------------
// Kernel 2: per (b, 16 l's) tile: LayerNorm + fused projection via bf16 WMMA.
// Writes out = hn@M + cb + residual (coalesced, [B,C,L]) and u = hn@Wu^T + bu.
// ---------------------------------------------------------------------------
__global__ __launch_bounds__(256) void k_phaseA(
    const float* __restrict__ x, const float* __restrict__ gamma,
    const float* __restrict__ beta,
    const __bf16* __restrict__ Mbf, const __bf16* __restrict__ Wubf,
    const float* __restrict__ cb, const float* __restrict__ bu,
    float* __restrict__ out, float* __restrict__ u)
{
    __shared__ float  xs[16][DIM];     // residual + output staging (8 KB)
    __shared__ __bf16 hnb[16][DIM];    // normalized activations, bf16 (4 KB)

    int tid = threadIdx.x;
    int b   = blockIdx.x >> 8;         // 256 tiles per batch
    int l0  = (blockIdx.x & 255) << 4;
    const float* xb = x + (size_t)b * DIM * LLEN + l0;

    #pragma unroll
    for (int t = 0; t < 8; ++t) {      // coalesced: 16 consecutive l per c
        int i = tid + t * 256;
        int c = i >> 4, li = i & 15;
        xs[li][c] = xb[(size_t)c * LLEN + li];
    }
    __syncthreads();

    // LayerNorm: 16 lanes per row, 8 channels each, half-wave shuffle reduce.
    int row = tid >> 4, sub = tid & 15;
    float s0 = 0.f, s1 = 0.f;
    #pragma unroll
    for (int q = 0; q < 8; ++q) {
        float v = xs[row][sub * 8 + q];
        s0 += v; s1 += v * v;
    }
    #pragma unroll
    for (int m = 1; m < 16; m <<= 1) {
        s0 += __shfl_xor(s0, m, 32);
        s1 += __shfl_xor(s1, m, 32);
    }
    float mu   = s0 * (1.f / DIM);
    float var  = s1 * (1.f / DIM) - mu * mu;
    float rstd = rsqrtf(var + LN_EPS);
    #pragma unroll
    for (int q = 0; q < 8; ++q) {
        int c = sub * 8 + q;
        hnb[row][c] = (__bf16)((xs[row][c] - mu) * rstd * gamma[c] + beta[c]);
    }
    __syncthreads();

    int w = tid >> 5, lane = tid & 31;
    int m16 = lane & 15;
    int hi8 = (lane & 16) ? 8 : 0;

    // A fragments (shared across all K sweeps): 8 packed-pair LDS dwords each.
    v16bf afr[4];
    #pragma unroll
    for (int kb = 0; kb < 4; ++kb) {
        v8u tmp;
        #pragma unroll
        for (int p = 0; p < 8; ++p) {
            int K = kb * 32 + kpair_base(p, lane);
            tmp[p] = *(const unsigned int*)&hnb[m16][K];
        }
        afr[kb] = __builtin_bit_cast(v16bf, tmp);
    }

    // pout block nb = w: K=128 as 4 chained 16x16x32 bf16 WMMAs.
    v8f acc = {};
    #pragma unroll
    for (int kb = 0; kb < 4; ++kb) {
        v16bf bw = *(const v16bf*)(Mbf + (size_t)((w * 4 + kb) * 32 + lane) * 16);
        acc = __builtin_amdgcn_wmma_f32_16x16x32_bf16(
                  false, afr[kb], false, bw, (short)0, acc, false, false);
    }
    #pragma unroll
    for (int k = 0; k < 8; ++k) {       // C layout: VGPR k -> row k or k+8
        int M = k + hi8;
        int c = w * 16 + m16;
        xs[M][c] = acc[k] + cb[c] + xs[M][c];   // + residual
    }

    if (w == 0) {                       // u = hn @ Wu^T + bu (one wave)
        v8f acc2 = {};
        #pragma unroll
        for (int kb = 0; kb < 4; ++kb) {
            v16bf bw = *(const v16bf*)(Wubf + (size_t)(kb * 32 + lane) * 16);
            acc2 = __builtin_amdgcn_wmma_f32_16x16x32_bf16(
                       false, afr[kb], false, bw, (short)0, acc2, false, false);
        }
        float* ub = u + (size_t)(b * LLEN + l0) * DST;
        #pragma unroll
        for (int k = 0; k < 8; ++k)
            ub[(k + hi8) * DST + m16] = acc2[k] + bu[m16];
    }
    __syncthreads();

    float* ob = out + (size_t)b * DIM * LLEN + l0;
    #pragma unroll
    for (int t = 0; t < 8; ++t) {       // coalesced store back
        int i = tid + t * 256;
        int c = i >> 4, li = i & 15;
        ob[(size_t)c * LLEN + li] = xs[li][c];
    }
}

// ---------------------------------------------------------------------------
// Kernel 3: the serial recurrence. 8 waves, 2 batches/wave (16-lane halves).
// State replicated per lane -> the 16-wide dot is lane-local; one bpermute
// broadcast of the new state per step. Critical path ~ mul-tree + tanh.
// ---------------------------------------------------------------------------
__global__ __launch_bounds__(256) void k_scan(
    const float* __restrict__ Amat, const float* __restrict__ u,
    float* __restrict__ states)
{
    int tid = threadIdx.x;
    int w = tid >> 5, lane = tid & 31;
    int s = lane & 15;
    int b = 2 * w + (lane >> 4);

    float Ar[16];
    #pragma unroll
    for (int k = 0; k < 16; ++k) Ar[k] = Amat[s * DST + k];  // A[s][k]
    float st[16];
    #pragma unroll
    for (int k = 0; k < 16; ++k) st[k] = 0.f;

    const float* ub = u      + (size_t)b * LLEN * DST + s;
    float*       sb = states + (size_t)b * LLEN * DST + s;

    for (int l = 0; l < LLEN; ++l) {
        float acc = ub[(size_t)l * DST];
        #pragma unroll
        for (int k = 0; k < 16; ++k) acc += st[k] * Ar[k];
        float nv = tanhf(acc);
        sb[(size_t)l * DST] = nv;
        #pragma unroll
        for (int k = 0; k < 16; ++k)          // broadcast within 16-lane half
            st[k] = __shfl(nv, (lane & 16) | k, 32);
    }
}

// ---------------------------------------------------------------------------
// Kernel 4: out += states @ WC^T via f32 WMMA (4 x 16x16x4), LDS transpose
// for coalesced read-modify-write of out[B,C,L].
// ---------------------------------------------------------------------------
__global__ __launch_bounds__(256) void k_phaseC(
    const float* __restrict__ states, const float* __restrict__ WC,
    float* __restrict__ out)
{
    __shared__ float stg[8][16][16];   // per-wave staging (8 KB)
    int tid = threadIdx.x;
    int w = tid >> 5, lane = tid & 31;
    int b  = blockIdx.x >> 8;
    int l0 = (blockIdx.x & 255) << 4;
    int c0 = w * 16;
    int m16 = lane & 15;
    int hi  = (lane & 16) ? 1 : 0;

    const float* sB = states + (size_t)(b * LLEN + l0) * DST;
    v8f acc = {};
    #pragma unroll
    for (int kc = 0; kc < 4; ++kc) {
        int ka = kc * 4 + hi * 2;
        v2f a, bm;
        a.x  = sB[m16 * DST + ka];            // A[m][k] = states[l0+m][k]
        a.y  = sB[m16 * DST + ka + 1];
        bm.x = WC[(c0 + m16) * DST + ka];     // B[k][n] = WC[c0+n][k]
        bm.y = WC[(c0 + m16) * DST + ka + 1];
        acc = __builtin_amdgcn_wmma_f32_16x16x4_f32(
                  false, a, false, bm, (short)0, acc, false, false);
    }
    #pragma unroll
    for (int k = 0; k < 8; ++k)
        stg[w][k + hi * 8][m16] = acc[k];

    float* ob = out + ((size_t)b * DIM + c0) * LLEN + l0;
    #pragma unroll
    for (int it = 0; it < 8; ++it) {          // coalesced over l
        int idx = lane + it * 32;
        int n = idx >> 4, m = idx & 15;
        float* p = ob + (size_t)n * LLEN + m;
        *p = *p + stg[w][m][n];
    }
}

// ---------------------------------------------------------------------------
extern "C" void kernel_launch(void* const* d_in, const int* in_sizes, int n_in,
                              void* d_out, int out_size, void* d_ws, size_t ws_size,
                              hipStream_t stream) {
    const float* x     = (const float*)d_in[0];
    const float* Amat  = (const float*)d_in[1];
    const float* Bm    = (const float*)d_in[2];
    const float* Cm    = (const float*)d_in[3];
    const float* D     = (const float*)d_in[4];
    const float* W_in  = (const float*)d_in[5];
    const float* b_in  = (const float*)d_in[6];
    const float* W_out = (const float*)d_in[7];
    const float* b_out = (const float*)d_in[8];
    const float* gamma = (const float*)d_in[9];
    const float* beta  = (const float*)d_in[10];
    float* out = (float*)d_out;

    char* ws = (char*)d_ws;
    size_t off = 0;
    auto alloc = [&](size_t bytes) -> void* {
        void* p = ws + off;
        off += (bytes + 255) & ~(size_t)255;
        return p;
    };
    float*  u      = (float*)alloc((size_t)BSZ * LLEN * DST * 4);   // 4 MB
    float*  states = (float*)alloc((size_t)BSZ * LLEN * DST * 4);   // 4 MB
    float*  Wu     = (float*)alloc(DST * DIM * 4);
    float*  Mg     = (float*)alloc(DIM * DIM * 4);
    float*  WC     = (float*)alloc(DIM * DST * 4);
    float*  cb     = (float*)alloc(DIM * 4);
    float*  bu     = (float*)alloc(DST * 4);
    __bf16* Mbf    = (__bf16*)alloc(DIM * DIM * 2);
    __bf16* Wubf   = (__bf16*)alloc(DIM * DST * 2);

    k_pre<<<1, 256, 0, stream>>>(Bm, Cm, D, W_in, b_in, W_out, b_out,
                                 Wu, Mg, WC, cb, bu, Mbf, Wubf);
    int tiles = BSZ * (LLEN / 16);   // 4096
    k_phaseA<<<tiles, 256, 0, stream>>>(x, gamma, beta, Mbf, Wubf, cb, bu, out, u);
    k_scan<<<1, 256, 0, stream>>>(Amat, u, states);
    k_phaseC<<<tiles, 256, 0, stream>>>(states, WC, out);
}
